// CausalAttention_10299331575843
// MI455X (gfx1250) — matrix operand order repaired and checked by reading, not compiled, not run on hardware
//
#include <hip/hip_runtime.h>

// ---------------------------------------------------------------------------
// CDNA5 (gfx1250) causal multi-head attention, bf16 WMMA pipeline.
// B=4, S=2048, E=1024, H=16, hd=64.  wave32 only.
// ---------------------------------------------------------------------------

typedef __attribute__((ext_vector_type(16))) __bf16 v16bf;
typedef __attribute__((ext_vector_type(8)))  __bf16 v8bf;
typedef __attribute__((ext_vector_type(8)))  float  v8f;

#define LOG2E 1.44269504088896340736f

__device__ __forceinline__ v8f wmma_bf16(v16bf a, v16bf b, v8f c) {
  // D = A(16x32 bf16) * B(32x16 bf16) + C(16x16 f32)
  return __builtin_amdgcn_wmma_f32_16x16x32_bf16(
      /*neg_a=*/false, a, /*neg_b=*/false, b,
      /*c_mod=*/(short)0, c, /*reuse_a=*/false, /*reuse_b=*/false);
}

__device__ __forceinline__ v16bf cat16(v8bf lo, v8bf hi) {
  return __builtin_shufflevector(lo, hi, 0,1,2,3,4,5,6,7,8,9,10,11,12,13,14,15);
}

// A-fragment (16x32 bf16): lane holds row (lane&15); K chunks
// {koff..koff+15} where koff = kbase + 16*(lane>>4).
__device__ __forceinline__ v16bf load_a_frag(const __bf16* rowp, int koff) {
  v8bf lo = *(const v8bf*)(rowp + koff);
  v8bf hi = *(const v8bf*)(rowp + koff + 8);
  return cat16(lo, hi);
}

// ---------------------------------------------------------------------------
// fp32 -> bf16 conversion (grid-stride)
// ---------------------------------------------------------------------------
__global__ void cvt_f32_bf16(const float* __restrict__ s, __bf16* __restrict__ d, int n) {
  int i = blockIdx.x * blockDim.x + threadIdx.x;
  int stride = gridDim.x * blockDim.x;
  for (; i < n; i += stride) d[i] = (__bf16)s[i];
}

// ---------------------------------------------------------------------------
// QKV projection: C[8192,3072] = X[8192,1024] * W[3072,1024]^T + bias
// Scatters into Q,K [BH, S, 64] (Q pre-scaled by 0.125) and V^T [BH, 64, S].
// One wave computes a 64x64 tile (16 WMMA accumulators).
// ---------------------------------------------------------------------------
__global__ __launch_bounds__(256)
void qkv_gemm(const __bf16* __restrict__ X, const __bf16* __restrict__ W,
              const float* __restrict__ bias,
              __bf16* __restrict__ Qb, __bf16* __restrict__ Kb,
              __bf16* __restrict__ VTb) {
  const int lane = threadIdx.x & 31;
  const int wave = threadIdx.x >> 5;
  const int l16  = lane & 15;
  const int half = lane >> 4;
  const int gw   = blockIdx.x * 8 + wave;     // 6144 waves total
  const int m0   = (gw & 127) * 64;           // 128 M-tiles
  const int n0   = (gw >> 7) * 64;            // 48  N-tiles

  v8f acc[4][4];
#pragma unroll
  for (int i = 0; i < 4; ++i)
#pragma unroll
    for (int j = 0; j < 4; ++j) acc[i][j] = (v8f)0.0f;

  for (int k0 = 0; k0 < 1024; k0 += 32) {
    v16bf a[4], b[4];
#pragma unroll
    for (int i = 0; i < 4; ++i)
      a[i] = load_a_frag(X + (m0 + 16 * i + l16) * 1024, k0 + 16 * half);
#pragma unroll
    for (int j = 0; j < 4; ++j)
      b[j] = *(const v16bf*)(W + (n0 + 16 * j + l16) * 1024 + k0 + 16 * half);
#pragma unroll
    for (int i = 0; i < 4; ++i)
#pragma unroll
      for (int j = 0; j < 4; ++j)
        acc[i][j] = wmma_bf16(a[i], b[j], acc[i][j]);
  }

  // Epilogue: bias, Q-scale, scatter to head-major layouts (bf16).
#pragma unroll
  for (int j = 0; j < 4; ++j) {
    const int col   = n0 + 16 * j + l16;
    const int which = col >> 10;        // 0=Q 1=K 2=V (uniform per tile)
    const int hc    = col & 1023;
    const int h     = hc >> 6;
    const int d     = hc & 63;
    const float bv  = bias[col];
#pragma unroll
    for (int i = 0; i < 4; ++i) {
#pragma unroll
      for (int r = 0; r < 8; ++r) {
        const int row = m0 + 16 * i + r + 8 * half;   // global (b*S + s)
        const int bb  = row >> 11;
        const int ss  = row & 2047;
        float v = acc[i][j][r] + bv;
        if (which == 0) {
          v *= 0.125f;  // 1/sqrt(hd)
          Qb[(((bb << 4) + h) * 2048 + ss) * 64 + d] = (__bf16)v;
        } else if (which == 1) {
          Kb[(((bb << 4) + h) * 2048 + ss) * 64 + d] = (__bf16)v;
        } else {
          VTb[(((bb << 4) + h) * 64 + d) * 2048 + ss] = (__bf16)v;
        }
      }
    }
  }
}

// ---------------------------------------------------------------------------
// Flash attention (causal, online softmax). One wave owns 16 query rows of
// one (b,h); key blocks of 64. Scores/O accumulate in f32 via WMMA.
// P (C-layout) -> A-layout transpose goes through a per-wave LDS slab.
// ---------------------------------------------------------------------------
#define PROW 72   // padded LDS row stride (bf16 elems) to spread banks

__global__ __launch_bounds__(256)
void attn_kernel(const __bf16* __restrict__ Qb, const __bf16* __restrict__ Kb,
                 const __bf16* __restrict__ VTb, __bf16* __restrict__ AO) {
  __shared__ __bf16 pst[8][16 * PROW];

  const int lane = threadIdx.x & 31;
  const int wave = threadIdx.x >> 5;
  const int l16  = lane & 15;
  const int half = lane >> 4;

  const int task  = blockIdx.x * 8 + wave;   // 8192 tasks
  const int qt    = task & 127;
  const int bh    = task >> 7;
  const int qbase = qt << 4;

  const __bf16* Qh = Qb  + (size_t)bh * 2048 * 64;
  const __bf16* Kh = Kb  + (size_t)bh * 2048 * 64;
  const __bf16* Vh = VTb + (size_t)bh * 64 * 2048;

  // Q A-fragments for d=0..31 and d=32..63 (reused across all key blocks).
  v16bf aq[2];
  aq[0] = load_a_frag(Qh + (qbase + l16) * 64, 0 + 16 * half);
  aq[1] = load_a_frag(Qh + (qbase + l16) * 64, 32 + 16 * half);

  float m_r[8], l_r[8];
  v8f o[4];
#pragma unroll
  for (int r = 0; r < 8; ++r) { m_r[r] = -1e30f; l_r[r] = 0.0f; }
#pragma unroll
  for (int dj = 0; dj < 4; ++dj) o[dj] = (v8f)0.0f;

  const int kb_last = (qbase + 15) >> 6;
  for (int kb = 0; kb <= kb_last; ++kb) {
    const int kbase = kb << 6;

    // ---- scores S = Q * K^T (16 x 64), f32 ----
    v8f s[4];
#pragma unroll
    for (int j = 0; j < 4; ++j) s[j] = (v8f)0.0f;
#pragma unroll
    for (int ds = 0; ds < 2; ++ds) {
#pragma unroll
      for (int j = 0; j < 4; ++j) {
        v16bf bk = *(const v16bf*)(Kh + (kbase + 16 * j + l16) * 64 + 32 * ds + 16 * half);
        s[j] = wmma_bf16(aq[ds], bk, s[j]);
      }
    }

    // ---- causal mask (only the diagonal block needs it) ----
    if (kb == kb_last) {
#pragma unroll
      for (int j = 0; j < 4; ++j) {
        const int kcol = kbase + 16 * j + l16;
#pragma unroll
        for (int r = 0; r < 8; ++r) {
          const int qrow = qbase + r + 8 * half;
          if (kcol > qrow) s[j][r] = -1e30f;
        }
      }
    }

    // ---- online softmax: row max (reduce across 16-lane half) ----
    float corr[8];
#pragma unroll
    for (int r = 0; r < 8; ++r) {
      float t = fmaxf(fmaxf(s[0][r], s[1][r]), fmaxf(s[2][r], s[3][r]));
      t = fmaxf(t, __shfl_xor(t, 1));
      t = fmaxf(t, __shfl_xor(t, 2));
      t = fmaxf(t, __shfl_xor(t, 4));
      t = fmaxf(t, __shfl_xor(t, 8));
      const float nm = fmaxf(m_r[r], t);
      corr[r] = exp2f((m_r[r] - nm) * LOG2E);
      m_r[r]  = nm;
    }

    // ---- P = exp(S - m); row sums; rescale running state ----
#pragma unroll
    for (int j = 0; j < 4; ++j)
#pragma unroll
      for (int r = 0; r < 8; ++r)
        s[j][r] = exp2f((s[j][r] - m_r[r]) * LOG2E);

#pragma unroll
    for (int r = 0; r < 8; ++r) {
      float ps = s[0][r] + s[1][r] + s[2][r] + s[3][r];
      ps += __shfl_xor(ps, 1);
      ps += __shfl_xor(ps, 2);
      ps += __shfl_xor(ps, 4);
      ps += __shfl_xor(ps, 8);
      l_r[r] = l_r[r] * corr[r] + ps;
    }
#pragma unroll
    for (int dj = 0; dj < 4; ++dj)
#pragma unroll
      for (int r = 0; r < 8; ++r) o[dj][r] *= corr[r];

    // ---- P (C-layout) -> LDS -> A-layout bf16 fragments ----
    __bf16* pw = &pst[wave][0];
#pragma unroll
    for (int j = 0; j < 4; ++j)
#pragma unroll
      for (int r = 0; r < 8; ++r)
        pw[(r + 8 * half) * PROW + 16 * j + l16] = (__bf16)s[j][r];

    asm volatile("s_wait_dscnt 0" ::: "memory");   // DS stores visible to our loads

    v16bf ap[2];
    ap[0] = cat16(*(const v8bf*)(pw + l16 * PROW + 16 * half),
                  *(const v8bf*)(pw + l16 * PROW + 16 * half + 8));
    ap[1] = cat16(*(const v8bf*)(pw + l16 * PROW + 32 + 16 * half),
                  *(const v8bf*)(pw + l16 * PROW + 32 + 16 * half + 8));

    // ---- O += P * V  (V^T rows are contiguous in k) ----
#pragma unroll
    for (int ks = 0; ks < 2; ++ks) {
#pragma unroll
      for (int dj = 0; dj < 4; ++dj) {
        v16bf bv = *(const v16bf*)(Vh + (16 * dj + l16) * 2048 + kbase + 32 * ks + 16 * half);
        o[dj] = wmma_bf16(ap[ks], bv, o[dj]);
      }
    }
    // WAR on pst across iterations is safe: DS ops are in-order per wave.
  }

  // ---- finalize: O / l, store to attn-out [B*S, E] in bf16 ----
  float inv[8];
#pragma unroll
  for (int r = 0; r < 8; ++r) inv[r] = 1.0f / l_r[r];

  const int h = bh & 15;
  const int b = bh >> 4;
#pragma unroll
  for (int dj = 0; dj < 4; ++dj)
#pragma unroll
    for (int r = 0; r < 8; ++r) {
      const int row = qbase + r + 8 * half;
      AO[((size_t)(b * 2048 + row)) * 1024 + (h << 6) + 16 * dj + l16] =
          (__bf16)(o[dj][r] * inv[r]);
    }
}

// ---------------------------------------------------------------------------
// Output projection: out[8192,1024] = AO[8192,1024] * Wo[1024,1024]^T + bias
// ---------------------------------------------------------------------------
__global__ __launch_bounds__(256)
void out_gemm(const __bf16* __restrict__ A, const __bf16* __restrict__ W,
              const float* __restrict__ bias, float* __restrict__ out) {
  const int lane = threadIdx.x & 31;
  const int wave = threadIdx.x >> 5;
  const int l16  = lane & 15;
  const int half = lane >> 4;
  const int gw   = blockIdx.x * 8 + wave;   // 2048 waves
  const int m0   = (gw & 127) * 64;
  const int n0   = (gw >> 7) * 64;          // 16 N-tiles

  v8f acc[4][4];
#pragma unroll
  for (int i = 0; i < 4; ++i)
#pragma unroll
    for (int j = 0; j < 4; ++j) acc[i][j] = (v8f)0.0f;

  for (int k0 = 0; k0 < 1024; k0 += 32) {
    v16bf a[4], b[4];
#pragma unroll
    for (int i = 0; i < 4; ++i)
      a[i] = load_a_frag(A + (m0 + 16 * i + l16) * 1024, k0 + 16 * half);
#pragma unroll
    for (int j = 0; j < 4; ++j)
      b[j] = *(const v16bf*)(W + (n0 + 16 * j + l16) * 1024 + k0 + 16 * half);
#pragma unroll
    for (int i = 0; i < 4; ++i)
#pragma unroll
      for (int j = 0; j < 4; ++j)
        acc[i][j] = wmma_bf16(a[i], b[j], acc[i][j]);
  }

#pragma unroll
  for (int j = 0; j < 4; ++j) {
    const int col  = n0 + 16 * j + l16;
    const float bv = bias[col];
#pragma unroll
    for (int i = 0; i < 4; ++i)
#pragma unroll
      for (int r = 0; r < 8; ++r) {
        const int row = m0 + 16 * i + r + 8 * half;
        out[(size_t)row * 1024 + col] = acc[i][j][r] + bv;
      }
  }
}

// ---------------------------------------------------------------------------
// Host launcher
// ---------------------------------------------------------------------------
extern "C" void kernel_launch(void* const* d_in, const int* in_sizes, int n_in,
                              void* d_out, int out_size, void* d_ws, size_t ws_size,
                              hipStream_t stream) {
  const float* x     = (const float*)d_in[0];   // [4,2048,1024]
  const float* w_in  = (const float*)d_in[1];   // [3072,1024]
  const float* b_in  = (const float*)d_in[2];   // [3072]
  const float* w_out = (const float*)d_in[3];   // [1024,1024]
  const float* b_out = (const float*)d_in[4];   // [1024]
  float* out = (float*)d_out;                   // [4,2048,1024]

  char* ws = (char*)d_ws;
  __bf16* Xb = (__bf16*)(ws);                              //  16 MB  x bf16
  __bf16* Wi = (__bf16*)(ws + 16777216);                   //   6 MB  in_proj_w bf16
  __bf16* Wo = (__bf16*)(ws + 16777216 + 6291456);         //   2 MB  out_proj_w bf16
  __bf16* Qb = (__bf16*)(ws + 25165824);                   //  16 MB  Q [BH,S,64]
  __bf16* Kb = (__bf16*)(ws + 41943040);                   //  16 MB  K [BH,S,64]
  __bf16* VT = (__bf16*)(ws + 58720256);                   //  16 MB  V^T [BH,64,S]
  __bf16* AO = (__bf16*)(ws + 75497472);                   //  16 MB  attn out [BS,E]

  cvt_f32_bf16<<<2048, 256, 0, stream>>>(x,     Xb, 8388608);
  cvt_f32_bf16<<<1024, 256, 0, stream>>>(w_in,  Wi, 3145728);
  cvt_f32_bf16<<<512,  256, 0, stream>>>(w_out, Wo, 1048576);

  qkv_gemm   <<<768,  256, 0, stream>>>(Xb, Wi, b_in, Qb, Kb, VT);
  attn_kernel<<<1024, 256, 0, stream>>>(Qb, Kb, VT, AO);
  out_gemm   <<<256,  256, 0, stream>>>(AO, Wo, b_out, out);
}
